// CausalSelfAttention_55207509622867
// MI455X (gfx1250) — matrix operand order repaired and checked by reading, not compiled, not run on hardware
//
#include <hip/hip_runtime.h>
#include <hip/hip_bf16.h>

// ---------------------------------------------------------------------------
// CDNA5 (gfx1250) fused causal-window attention block.
// All matrix math via v_wmma_f32_16x16x32_f16 (f16 in, f32 accumulate).
// Wave32 everywhere. Fragment layouts used in this file:
//   A (16x32 f16): lane l -> row m=l%16; half h=l/16 holds K [16h..16h+7] in
//                  elements 0..7 and K [16h+8..16h+15] in elements 8..15.
//   B (32x16 f16): lane l -> col n=l%16; half h holds K [16h..16h+15] in
//                  elements 0..15 (contiguous run of 16).
//   C/D (16x16 f32): lane l -> col n=l%16; VGPR r -> row m=r+8*(l/16).
// ---------------------------------------------------------------------------

typedef _Float16 h8  __attribute__((ext_vector_type(8)));
typedef _Float16 h16 __attribute__((ext_vector_type(16)));
typedef float    f8  __attribute__((ext_vector_type(8)));

#define B_   4
#define T_   2048
#define C_   1024
#define H_   16
#define HD_  64
#define M_   (B_ * T_)        // 8192 rows (b,t)

static __device__ __forceinline__ f8 fzero() {
    f8 z;
#pragma unroll
    for (int i = 0; i < 8; ++i) z[i] = 0.0f;
    return z;
}

static __device__ __forceinline__ f8 wmma16(h16 a, h16 b, f8 c) {
    return __builtin_amdgcn_wmma_f32_16x16x32_f16(
        /*neg_a=*/false, a, /*neg_b=*/false, b,
        /*c_mod=*/(short)0, c, /*reuse_a=*/false, /*reuse_b=*/false);
}

// A fragment from a row-major (M,K) f16 matrix; base -> &A[m][k0].
static __device__ __forceinline__ h16 load_a_h(const _Float16* base, int hl) {
    h16 r;
    const h8* plo = (const h8*)(base + hl * 16);
    const h8* phi = (const h8*)(base + hl * 16 + 8);
    h8 lo = *plo, hi = *phi;
#pragma unroll
    for (int i = 0; i < 8; ++i) { r[i] = lo[i]; r[8 + i] = hi[i]; }
    return r;
}

// B fragment: stored matrix is row-major (N,K); lane's column n of B == stored
// row n, contiguous over K. base -> &Bstored[n][k0].
static __device__ __forceinline__ h16 load_b_h(const _Float16* base, int hl) {
    h16 r;
    const h8* p = (const h8*)(base + hl * 16);
    h8 lo = p[0], hi = p[1];
#pragma unroll
    for (int i = 0; i < 8; ++i) { r[i] = lo[i]; r[8 + i] = hi[i]; }
    return r;
}

// ---------------------------------------------------------------------------
// Kernel 0: one-shot f32 -> f16 conversion (x, qkv_weight, c_proj_w).
// 8 elements / thread, b128 in / b128 out. n must be a multiple of 8.
// ---------------------------------------------------------------------------
__global__ __launch_bounds__(256) void cvt_f16_kernel(
    const float* __restrict__ in, _Float16* __restrict__ out, int n8) {
    const int i = blockIdx.x * 256 + threadIdx.x;
    if (i >= n8) return;
    const size_t base = (size_t)i * 8;
    h8 o;
#pragma unroll
    for (int j = 0; j < 8; ++j) o[j] = (_Float16)in[base + j];
    *(h8*)(out + base) = o;
}

// ---------------------------------------------------------------------------
// Kernel 1/4: C[M,N] = A[M,K] * Bstored[N,K]^T (optionally + bias), f32 out.
// A and Bstored both pre-converted f16. Block: 256 thr = 8 waves, tile
// 64(M) x 256(N); wave tile 32x64 = 2x4 WMMA. Inner loop: b128 loads + WMMA.
// ---------------------------------------------------------------------------
template <bool USE_BIAS>
__global__ __launch_bounds__(256) void gemm_nt_kernel(
    const _Float16* __restrict__ Ap, const _Float16* __restrict__ Bp,
    const float* __restrict__ bias, float* __restrict__ Cp,
    int M, int N, int K, int lda, int ldb, int ldc) {
    const int l  = threadIdx.x & 31;
    const int w  = threadIdx.x >> 5;
    const int lm = l & 15;
    const int hl = l >> 4;
    const int wm = w & 1;        // 2 wave slots along M  (2*32 = 64)
    const int wn = w >> 1;       // 4 wave slots along N  (4*64 = 256)
    const int bm = blockIdx.x * 64;
    const int bn = blockIdx.y * 256;

    int arow[2], bcol[4];
#pragma unroll
    for (int f = 0; f < 2; ++f) arow[f] = bm + wm * 32 + f * 16 + lm;
#pragma unroll
    for (int f = 0; f < 4; ++f) bcol[f] = bn + wn * 64 + f * 16 + lm;

    f8 acc[2][4];
#pragma unroll
    for (int i = 0; i < 2; ++i)
#pragma unroll
        for (int j = 0; j < 4; ++j) acc[i][j] = fzero();

    for (int k0 = 0; k0 < K; k0 += 32) {
        h16 a[2], b[4];
#pragma unroll
        for (int f = 0; f < 2; ++f)
            a[f] = load_a_h(Ap + (size_t)arow[f] * lda + k0, hl);
#pragma unroll
        for (int f = 0; f < 4; ++f)
            b[f] = load_b_h(Bp + (size_t)bcol[f] * ldb + k0, hl);
#pragma unroll
        for (int i = 0; i < 2; ++i)
#pragma unroll
            for (int j = 0; j < 4; ++j)
                acc[i][j] = wmma16(a[i], b[j], acc[i][j]);
    }

#pragma unroll
    for (int fm = 0; fm < 2; ++fm) {
        const int rowb = bm + wm * 32 + fm * 16 + 8 * hl;
#pragma unroll
        for (int fn = 0; fn < 4; ++fn) {
            const int   col = bcol[fn];
            const float bv  = USE_BIAS ? bias[col] : 0.0f;
#pragma unroll
            for (int r = 0; r < 8; ++r)
                Cp[(size_t)(rowb + r) * ldc + col] = acc[fm][fn][r] + bv;
        }
    }
}

// ---------------------------------------------------------------------------
// Kernel 2/4: value mix + RMS-norm(q,k) + rotary. One wave per (b,h,t) row;
// lane d handles elements d and d+32 (the rotary pair). 8 waves per block.
// ---------------------------------------------------------------------------
__global__ __launch_bounds__(256) void fuse_norm_rope_kernel(
    const float* __restrict__ qkv, const float* __restrict__ v1,
    const float* __restrict__ lamb,
    _Float16* __restrict__ qh, _Float16* __restrict__ kh,
    _Float16* __restrict__ vh, float* __restrict__ val_out) {
    const int l   = threadIdx.x & 31;
    const int w   = threadIdx.x >> 5;
    const int rid = blockIdx.x * 8 + w;          // (b*H + h)*T + t
    const int b   = rid >> 15;                   // / (H_*T_)
    const int h   = (rid >> 11) & (H_ - 1);      // / T_ % H_
    const int t   = rid & (T_ - 1);

    const float lam = lamb[0];

    const size_t qkvoff = (size_t)(b * T_ + t) * 3072 + h * 64;
    float q1 = qkv[qkvoff + l],        q2 = qkv[qkvoff + l + 32];
    float k1 = qkv[qkvoff + 1024 + l], k2 = qkv[qkvoff + 1024 + l + 32];
    float va = qkv[qkvoff + 2048 + l], vb = qkv[qkvoff + 2048 + l + 32];

    const size_t v1off = (size_t)(b * T_ + t) * C_ + h * 64;
    const float w1 = v1[v1off + l], w2 = v1[v1off + l + 32];
    const float mix1 = (1.0f - lam) * va + lam * w1;
    const float mix2 = (1.0f - lam) * vb + lam * w2;

    const size_t ho = (size_t)((b * H_ + h) * T_ + t) * HD_;
    val_out[ho + l]      = mix1;
    val_out[ho + l + 32] = mix2;
    vh[ho + l]      = (_Float16)mix1;
    vh[ho + l + 32] = (_Float16)mix2;

    // RMS norm over the 64-wide row (wave reduction).
    float ssq = q1 * q1 + q2 * q2;
    float ssk = k1 * k1 + k2 * k2;
#pragma unroll
    for (int off = 16; off >= 1; off >>= 1) {
        ssq += __shfl_xor(ssq, off, 32);
        ssk += __shfl_xor(ssk, off, 32);
    }
    const float invq = rsqrtf(ssq * (1.0f / 64.0f) + 1e-6f);
    const float invk = rsqrtf(ssk * (1.0f / 64.0f) + 1e-6f);
    q1 *= invq; q2 *= invq; k1 *= invk; k2 *= invk;

    // rotary: inv_freq[i] = (1/1024)^(i/15) for i<16, else 0  => 2^(-10*i/15)
    const float fr = (l < 16) ? exp2f(-10.0f * (float)l * (1.0f / 15.0f)) : 0.0f;
    const float th = (float)t * fr;
    float s, c;
    sincosf(th, &s, &c);

    qh[ho + l]      = (_Float16)( q1 * c + q2 * s);
    qh[ho + l + 32] = (_Float16)(-q1 * s + q2 * c);
    kh[ho + l]      = (_Float16)( k1 * c + k2 * s);
    kh[ho + l + 32] = (_Float16)(-k1 * s + k2 * c);
}

// ---------------------------------------------------------------------------
// Kernel 3/4: sliding-window causal attention. One wave per 16-query tile of
// one (b,h). Two-pass softmax: pass1 row max; pass2 exp + P*V (P transposed
// through LDS into A-fragment layout). 4 waves / block. All masking is
// branchless (cndmask), no exec divergence in the hot loops.
// ---------------------------------------------------------------------------
__global__ __launch_bounds__(128) void attn_kernel(
    const _Float16* __restrict__ qh, const _Float16* __restrict__ kh,
    const _Float16* __restrict__ vh, const int* __restrict__ winp,
    _Float16* __restrict__ aout) {
    __shared__ __align__(16) _Float16 plds[4][16][32];   // per-wave P tile

    const int l  = threadIdx.x & 31;
    const int w  = threadIdx.x >> 5;
    const int lm = l & 15;
    const int hl = l >> 4;

    const int tile = blockIdx.x * 4 + w;         // < B*H*(T/16) = 8192
    const int tq   = tile & (T_ / 16 - 1);
    const int bh   = tile >> 7;                  // b*H + h
    const int b    = bh >> 4;
    const int h    = bh & (H_ - 1);
    const int t0   = tq << 4;
    const unsigned win = (unsigned)winp[0];

    const _Float16* qbase = qh + (size_t)bh * T_ * HD_;
    const _Float16* kbase = kh + (size_t)bh * T_ * HD_;
    const _Float16* vbase = vh + (size_t)bh * T_ * HD_;

    // Q fragments (16 queries x 64), loaded once.
    h16 aq[2];
    {
        const _Float16* qr = qbase + (size_t)(t0 + lm) * HD_;
        aq[0] = load_a_h(qr, hl);
        aq[1] = load_a_h(qr + 32, hl);
    }

    int js = t0 - (int)win + 1;
    if (js < 0) js = 0;
    js &= ~31;                                   // 32-key groups; mask fixes slop

    const float scale = 0.125f;                  // 1/sqrt(64)
    const float NEGBIG = -1e30f;

    // ---- pass 1: row max (branchless) ----
    float rmax[8];
#pragma unroll
    for (int r = 0; r < 8; ++r) rmax[r] = NEGBIG;

    for (int j0 = js; j0 < t0 + 16; j0 += 32) {
#pragma unroll
        for (int sub = 0; sub < 2; ++sub) {
            const int jb  = j0 + sub * 16;
            int key = jb + lm; if (key > T_ - 1) key = T_ - 1;
            const _Float16* kr = kbase + (size_t)key * HD_;
            h16 bk0 = load_b_h(kr, hl);
            h16 bk1 = load_b_h(kr + 32, hl);
            f8 s = fzero();
            s = wmma16(aq[0], bk0, s);
            s = wmma16(aq[1], bk1, s);
            const int j = jb + lm;               // this lane's key column
#pragma unroll
            for (int r = 0; r < 8; ++r) {
                const int i = t0 + r + 8 * hl;
                // allowed <=> 0 <= i-j < win  (single unsigned compare)
                const bool  ok = (unsigned)(i - j) < win;
                const float sv = ok ? s[r] * scale : NEGBIG;
                rmax[r] = fmaxf(rmax[r], sv);
            }
        }
    }
#pragma unroll
    for (int r = 0; r < 8; ++r)
#pragma unroll
        for (int off = 8; off >= 1; off >>= 1)
            rmax[r] = fmaxf(rmax[r], __shfl_xor(rmax[r], off, 32));

    // ---- pass 2: exp, row sums, P*V ----
    float rsum[8];
#pragma unroll
    for (int r = 0; r < 8; ++r) rsum[r] = 0.0f;
    f8 acc[4];
#pragma unroll
    for (int ch = 0; ch < 4; ++ch) acc[ch] = fzero();

    for (int j0 = js; j0 < t0 + 16; j0 += 32) {
#pragma unroll
        for (int sub = 0; sub < 2; ++sub) {
            const int jb  = j0 + sub * 16;
            int key = jb + lm; if (key > T_ - 1) key = T_ - 1;
            const _Float16* kr = kbase + (size_t)key * HD_;
            h16 bk0 = load_b_h(kr, hl);
            h16 bk1 = load_b_h(kr + 32, hl);
            f8 s = fzero();
            s = wmma16(aq[0], bk0, s);
            s = wmma16(aq[1], bk1, s);
            const int j = jb + lm;
#pragma unroll
            for (int r = 0; r < 8; ++r) {
                const int i = t0 + r + 8 * hl;
                const bool ok = (unsigned)(i - j) < win;
                // select the *argument*: masked lanes get exp(-1e30) == 0,
                // no control flow, no inf*0.
                const float arg = ok ? (s[r] * scale - rmax[r]) : NEGBIG;
                const float p   = __expf(arg);
                rsum[r] += p;
                plds[w][r + 8 * hl][sub * 16 + lm] = (_Float16)p;  // transpose
            }
        }
        // make this wave's LDS stores visible before fragment re-read
        asm volatile("s_wait_dscnt 0x0" ::: "memory");

        h16 pa = load_a_h(&plds[w][lm][0], hl);   // P in A-frag layout, K=32 keys
#pragma unroll
        for (int ch = 0; ch < 4; ++ch) {
            h16 bv;
#pragma unroll
            for (int i = 0; i < 16; ++i) {
                int key = j0 + hl * 16 + i;
                if (key > T_ - 1) key = T_ - 1;   // masked rows contribute p=0
                bv[i] = vbase[(size_t)key * HD_ + ch * 16 + lm];
            }
            acc[ch] = wmma16(pa, bv, acc[ch]);
        }
    }

#pragma unroll
    for (int r = 0; r < 8; ++r)
#pragma unroll
        for (int off = 8; off >= 1; off >>= 1)
            rsum[r] += __shfl_xor(rsum[r], off, 32);

    // write (B,T,C) f16 for the c_proj GEMM
#pragma unroll
    for (int r = 0; r < 8; ++r) {
        const float rinv = 1.0f / rsum[r];
        const int   t    = t0 + r + 8 * hl;
        const size_t rowoff = (size_t)(b * T_ + t) * C_ + h * 64;
#pragma unroll
        for (int ch = 0; ch < 4; ++ch)
            aout[rowoff + ch * 16 + lm] = (_Float16)(acc[ch][r] * rinv);
    }
}

// ---------------------------------------------------------------------------
// Launch: cvt -> qkv GEMM -> mix/norm/rope -> attention -> c_proj GEMM.
// ---------------------------------------------------------------------------
extern "C" void kernel_launch(void* const* d_in, const int* in_sizes, int n_in,
                              void* d_out, int out_size, void* d_ws, size_t ws_size,
                              hipStream_t stream) {
    (void)in_sizes; (void)n_in; (void)out_size; (void)ws_size;

    const float* x    = (const float*)d_in[0];   // (B,T,C)
    const float* v1   = (const float*)d_in[1];   // (B,T,C)
    const float* qkvw = (const float*)d_in[2];   // (3,C,C) == (3072,1024)
    const float* cpw  = (const float*)d_in[3];   // (C,C)
    const float* cpb  = (const float*)d_in[4];   // (C,)
    const float* lamb = (const float*)d_in[5];   // (1,)
    const int*   win  = (const int*)d_in[6];     // scalar 512

    char* ws = (char*)d_ws;
    size_t off = 0;
    float* qkv_ws = (float*)(ws + off); off += (size_t)M_ * 3072 * sizeof(float);
    _Float16* qh  = (_Float16*)(ws + off); off += (size_t)M_ * C_ * sizeof(_Float16);
    _Float16* kh  = (_Float16*)(ws + off); off += (size_t)M_ * C_ * sizeof(_Float16);
    _Float16* vh  = (_Float16*)(ws + off); off += (size_t)M_ * C_ * sizeof(_Float16);
    _Float16* at  = (_Float16*)(ws + off); off += (size_t)M_ * C_ * sizeof(_Float16);
    _Float16* x16 = (_Float16*)(ws + off); off += (size_t)M_ * C_ * sizeof(_Float16);
    _Float16* qw16 = (_Float16*)(ws + off); off += (size_t)3 * C_ * C_ * sizeof(_Float16);
    _Float16* cw16 = (_Float16*)(ws + off);

    float* y       = (float*)d_out;                               // (B,T,C)
    float* val_out = y + (size_t)M_ * C_;                         // (B,H,T,hd)

    // 0) one-shot f32 -> f16 of GEMM operands
    {
        const int nx = M_ * C_ / 8, nq = 3 * C_ * C_ / 8, nc = C_ * C_ / 8;
        cvt_f16_kernel<<<(nx + 255) / 256, 256, 0, stream>>>(x, x16, nx);
        cvt_f16_kernel<<<(nq + 255) / 256, 256, 0, stream>>>(qkvw, qw16, nq);
        cvt_f16_kernel<<<(nc + 255) / 256, 256, 0, stream>>>(cpw, cw16, nc);
    }

    // 1) qkv = x @ qkv_weight^T : M=8192, N=3072, K=1024
    gemm_nt_kernel<false><<<dim3(M_ / 64, 3072 / 256), 256, 0, stream>>>(
        x16, qw16, nullptr, qkv_ws, M_, 3072, C_, C_, C_, 3072);

    // 2) value mix + rmsnorm + rope : one wave per (b,h,t) row
    fuse_norm_rope_kernel<<<(B_ * H_ * T_) / 8, 256, 0, stream>>>(
        qkv_ws, v1, lamb, qh, kh, vh, val_out);

    // 3) windowed causal attention : one wave per 16-query tile
    attn_kernel<<<(B_ * H_ * (T_ / 16)) / 4, 128, 0, stream>>>(
        qh, kh, vh, win, at);

    // 4) y = attn @ c_proj_w^T + b : M=8192, N=1024, K=1024
    gemm_nt_kernel<true><<<dim3(M_ / 64, C_ / 256), 256, 0, stream>>>(
        at, cw16, cpb, y, M_, C_, C_, C_, C_, C_);
}
